// HilbertAttentionCore_46188078301816
// MI455X (gfx1250) — matrix operand order repaired and checked by reading, not compiled, not run on hardware
//
#include <hip/hip_runtime.h>
#include <hip/hip_bf16.h>
#include <math.h>

typedef __attribute__((ext_vector_type(16))) _Float16 v16h;
typedef __attribute__((ext_vector_type(8)))  _Float16 v8h;
typedef __attribute__((ext_vector_type(8)))  float    v8f;
typedef __attribute__((ext_vector_type(4)))  float    v4f;

#define NB 2
#define NH 16
#define NM 8192
#define ND 64
#define SEG 256
#define NKEY 128            // SEG / dilation_rate
#define NWAVE 16
#define KSTR 72             // K LDS row stride (halves), padded: conflict-free
#define VSTR 136            // V^T LDS row stride (halves), padded
#define PSTR 136            // P LDS row stride (halves), padded

// Combine two 16B LDS loads into a 16-half WMMA fragment register block.
__device__ __forceinline__ v16h ld16h(const _Float16* p) {
  v8h a = *(const v8h*)p;
  v8h b = *(const v8h*)(p + 8);
  return __builtin_shufflevector(a, b, 0,1,2,3,4,5,6,7,8,9,10,11,12,13,14,15);
}

__global__ __launch_bounds__(512, 1) void hilbert_attn_wmma(
    const float* __restrict__ q, const float* __restrict__ k,
    const float* __restrict__ v, const int* __restrict__ hm,
    float* __restrict__ out)
{
  __shared__ __align__(16) _Float16 sK [NKEY * KSTR];     // [key][d]   row-major f16
  __shared__ __align__(16) _Float16 sVT[ND   * VSTR];     // [d][key]   transposed f16
  __shared__ __align__(16) _Float16 sP [NWAVE * 16 * PSTR]; // per-wave P rows

  const int S    = NM / SEG;            // 32 segments
  const int seg  = blockIdx.x % S;
  const int bh   = blockIdx.x / S;      // b*H + h
  const int tid  = threadIdx.x;
  const int wave = tid >> 5;
  const int lane = tid & 31;
  const int g    = lane >> 4;           // half-wave id
  const int ln   = lane & 15;

  const size_t baseBH  = (size_t)bh * NM * ND;
  const int    segBase = seg * SEG;

  // ---------------- stage K (row-major) and V (transposed) into LDS ---------------
  {
    const int row = tid >> 2;                 // selected-key index 0..127
    const int c0  = (tid & 3) << 4;           // 16-col chunk of D
    const int src = hm[segBase + 2 * row];    // dilation_rate = 2 gather
    const float* kp = k + baseBH + (size_t)src * ND + c0;
    const float* vp = v + baseBH + (size_t)src * ND + c0;

    float kf[16], vf[16];
    #pragma unroll
    for (int i = 0; i < 4; ++i) {
      v4f kk = *(const v4f*)(kp + 4 * i);
      v4f vv = *(const v4f*)(vp + 4 * i);
      #pragma unroll
      for (int j = 0; j < 4; ++j) { kf[4*i+j] = kk[j]; vf[4*i+j] = vv[j]; }
    }
    v8h h0, h1;
    #pragma unroll
    for (int i = 0; i < 8; ++i) { h0[i] = (_Float16)kf[i]; h1[i] = (_Float16)kf[8+i]; }
    *(v8h*)&sK[row * KSTR + c0]     = h0;
    *(v8h*)&sK[row * KSTR + c0 + 8] = h1;
    #pragma unroll
    for (int i = 0; i < 16; ++i)              // transpose V on the way in
      sVT[(size_t)(c0 + i) * VSTR + row] = (_Float16)vf[i];
  }

  // ---------------- load this wave's Q fragments (A-matrix 16x32 f16 layout) -------
  // lane holds row m = ln; chunk c: halves 0..7 -> d = 32c+16g+0..7,
  //                               halves 8..15 -> d = 32c+16g+8..15
  const int qsrc = hm[segBase + wave * 16 + ln];
  const float* qp = q + baseBH + (size_t)qsrc * ND;
  v16h aQ[2];
  #pragma unroll
  for (int c = 0; c < 2; ++c) {
    const float* p0 = qp + 32 * c + 16 * g;
    const float* p1 = qp + 32 * c + 16 * g + 8;
    v4f q0 = *(const v4f*)(p0);     v4f q1 = *(const v4f*)(p0 + 4);
    v4f q2 = *(const v4f*)(p1);     v4f q3 = *(const v4f*)(p1 + 4);
    #pragma unroll
    for (int j = 0; j < 4; ++j) {
      aQ[c][j]      = (_Float16)(q0[j] * 0.125f);   // fold scale = 1/sqrt(64)
      aQ[c][4 + j]  = (_Float16)(q1[j] * 0.125f);
      aQ[c][8 + j]  = (_Float16)(q2[j] * 0.125f);
      aQ[c][12 + j] = (_Float16)(q3[j] * 0.125f);
    }
  }

  __syncthreads();

  // ---------------- scores: S(16x128) = Q(16x64) * K^T, 8 tiles of 16 keys ---------
  v8f st[8];
  #pragma unroll
  for (int t = 0; t < 8; ++t) {
    // B-fragment: lane holds B column n = ln (key t*16+ln), k-values d = 32c+16g+0..15
    const _Float16* kr = &sK[(size_t)(t * 16 + ln) * KSTR + 16 * g];
    v16h b0 = ld16h(kr);
    v16h b1 = ld16h(kr + 32);
    v8f c = {};
    c = __builtin_amdgcn_wmma_f32_16x16x32_f16(false, aQ[0], false, b0, (short)0, c, false, false);
    c = __builtin_amdgcn_wmma_f32_16x16x32_f16(false, aQ[1], false, b1, (short)0, c, false, false);
    st[t] = c;
  }

  // ---------------- softmax over 128 keys (row m = 8g + r, cols across 16 lanes) ---
  float rowmax[8], rowsum[8];
  #pragma unroll
  for (int r = 0; r < 8; ++r) {
    float m = st[0][r];
    #pragma unroll
    for (int t = 1; t < 8; ++t) m = fmaxf(m, st[t][r]);
    #pragma unroll
    for (int msk = 1; msk <= 8; msk <<= 1) m = fmaxf(m, __shfl_xor(m, msk, 32));
    rowmax[r] = m;
  }
  #pragma unroll
  for (int r = 0; r < 8; ++r) {
    float s = 0.0f;
    #pragma unroll
    for (int t = 0; t < 8; ++t) {
      float e = __expf(st[t][r] - rowmax[r]);
      st[t][r] = e;
      s += e;
    }
    #pragma unroll
    for (int msk = 1; msk <= 8; msk <<= 1) s += __shfl_xor(s, msk, 32);
    rowsum[r] = s;
  }

  // ---------------- write normalized P (f16) to per-wave LDS, row-major -----------
  _Float16* pw = &sP[(size_t)wave * 16 * PSTR];
  #pragma unroll
  for (int r = 0; r < 8; ++r) {
    const float inv = 1.0f / rowsum[r];
    const int m = 8 * g + r;
    #pragma unroll
    for (int t = 0; t < 8; ++t)
      pw[(size_t)m * PSTR + t * 16 + ln] = (_Float16)(st[t][r] * inv);
  }

  // ---------------- out(16x64) = P(16x128) * V(128x64), fp32 accum -----------------
  v8f acc[4];
  #pragma unroll
  for (int t = 0; t < 4; ++t) acc[t] = (v8f){};

  #pragma unroll
  for (int c = 0; c < 4; ++c) {                 // key chunks of 32
    // A-fragment: row m = ln of P; k = 32c+16g+0..15
    const _Float16* pr = &pw[(size_t)ln * PSTR + 32 * c];
    v8h a0 = *(const v8h*)(pr + 16 * g);
    v8h a1 = *(const v8h*)(pr + 16 * g + 8);
    v16h a = __builtin_shufflevector(a0, a1, 0,1,2,3,4,5,6,7,8,9,10,11,12,13,14,15);
    #pragma unroll
    for (int t = 0; t < 4; ++t) {               // d tiles of 16
      // B-fragment: column n = d-col (t*16+ln) of V == row of V^T; contiguous keys
      const _Float16* vr = &sVT[(size_t)(t * 16 + ln) * VSTR + 32 * c + 16 * g];
      v16h bv = ld16h(vr);
      acc[t] = __builtin_amdgcn_wmma_f32_16x16x32_f16(false, a, false, bv, (short)0, acc[t], false, false);
    }
  }

  // ---------------- store output (permuted order, per C/D 16x16 f32 layout) -------
  float* op = out + baseBH + (size_t)(segBase + wave * 16) * ND;
  #pragma unroll
  for (int t = 0; t < 4; ++t) {
    #pragma unroll
    for (int r = 0; r < 8; ++r) {
      const int m = 8 * g + r;
      op[(size_t)m * ND + t * 16 + ln] = acc[t][r];
    }
  }
}

extern "C" void kernel_launch(void* const* d_in, const int* in_sizes, int n_in,
                              void* d_out, int out_size, void* d_ws, size_t ws_size,
                              hipStream_t stream) {
  const float* q  = (const float*)d_in[0];
  const float* k  = (const float*)d_in[1];
  const float* v  = (const float*)d_in[2];
  const int*   hm = (const int*)d_in[3];
  // d_in[4] = segment_size (256), d_in[5] = dilation_rate (2): baked into kernel.
  float* out = (float*)d_out;

  const int blocks = NB * NH * (NM / SEG);   // 1024 (b,h,segment) tiles
  hilbert_attn_wmma<<<blocks, 512, 0, stream>>>(q, k, v, hm, out);
}
